// LSTELinear_15058155339826
// MI455X (gfx1250) — matrix operand order repaired and checked by reading, not compiled, not run on hardware
//
#include <hip/hip_runtime.h>

// Problem constants (from reference)
#define IN_F   4096   // K
#define OUT_F  4096   // N
#define TOKENS 8192   // M
#define GS     128

// GEMM tiling (double-buffered LDS: 2*(128+64)*72*2 = 55296 B < 64 KB)
#define BM 128
#define BN 64
#define BK 64
#define LDSK (BK + 8)   // padded row stride (halves): 144 B, 16B-aligned, breaks bank cycling

#define XSTAGE_HALVES (BM * LDSK)                    // 9216 halves = 18432 B
#define WSTAGE_HALVES (BN * LDSK)                    // 4608 halves =  9216 B
#define STAGE_HALVES  (XSTAGE_HALVES + WSTAGE_HALVES)  // 13824 halves = 27648 B
#define STAGE_BYTES   (STAGE_HALVES * 2)

typedef __attribute__((ext_vector_type(16))) __bf16 v16bf;
typedef __attribute__((ext_vector_type(8)))  float  v8f;

union Frag {
    v16bf v;
    uint4 q[2];
};

__device__ __forceinline__ unsigned short f2bf(float f) {
    union { float f; unsigned u; } c; c.f = f;
    unsigned u = c.u + 0x7FFFu + ((c.u >> 16) & 1u);  // round-to-nearest-even
    return (unsigned short)(u >> 16);
}
__device__ __forceinline__ unsigned pack2bf(float a, float b) {
    return (unsigned)f2bf(a) | ((unsigned)f2bf(b) << 16);
}

// ---------------------------------------------------------------------------
// Pass 1: x f32 -> bf16 (one-time; 8 elements / thread)
// ---------------------------------------------------------------------------
__global__ __launch_bounds__(256)
void cvt_x_bf16(const float* __restrict__ x, unsigned short* __restrict__ xb) {
    size_t i = ((size_t)blockIdx.x * 256 + threadIdx.x) * 8;
    float4 a = *(const float4*)(x + i);
    float4 b = *(const float4*)(x + i + 4);
    uint4 o;
    o.x = pack2bf(a.x, a.y);
    o.y = pack2bf(a.z, a.w);
    o.z = pack2bf(b.x, b.y);
    o.w = pack2bf(b.z, b.w);
    *(uint4*)(xb + i) = o;
}

// ---------------------------------------------------------------------------
// Pass 2: ternary int8 * group-scale -> bf16 W (one-time; 16 weights / thread)
// ---------------------------------------------------------------------------
__global__ __launch_bounds__(256)
void dequant_w_bf16(const signed char* __restrict__ t, const float* __restrict__ s,
                    unsigned short* __restrict__ wb) {
    size_t i = ((size_t)blockIdx.x * 256 + threadIdx.x) * 16;
    int4 w = *(const int4*)(t + i);
    float sc = s[i / GS];
    const signed char* b = (const signed char*)&w;
    unsigned q[8];
    #pragma unroll
    for (int j = 0; j < 8; ++j)
        q[j] = pack2bf((float)b[2 * j] * sc, (float)b[2 * j + 1] * sc);
    *(uint4*)(wb + i)     = make_uint4(q[0], q[1], q[2], q[3]);
    *(uint4*)(wb + i + 8) = make_uint4(q[4], q[5], q[6], q[7]);
}

// ---------------------------------------------------------------------------
// Pass 3: bf16 GEMM with async global->LDS DMA. Double-buffered LDS with a
// RUNTIME stage offset + unroll(1) so accumulators stay pinned in place
// (no v_mov_b64 rotation, no WMMA->VALU hazard NOPs).
// ---------------------------------------------------------------------------
__global__ __launch_bounds__(256)
void gemm_bf16_wmma_async(const unsigned short* __restrict__ xb,
                          const unsigned short* __restrict__ wb,
                          float* __restrict__ out) {
    // [stage0: Xtile | Wtile][stage1: Xtile | Wtile]
    __shared__ __align__(16) unsigned short lds[2 * STAGE_HALVES];

    const int tid   = threadIdx.x;
    const int lane  = tid & 31;
    const int wid   = tid >> 5;
    const int waveM = wid >> 1;   // 0..3  (32 rows each)
    const int waveN = wid & 1;    // 0..1  (32 cols each)
    const int blockN = blockIdx.x * BN;
    const int blockM = blockIdx.y * BM;

    v8f acc[2][2];
    #pragma unroll
    for (int mi = 0; mi < 2; ++mi)
        #pragma unroll
        for (int ni = 0; ni < 2; ++ni)
            #pragma unroll
            for (int r = 0; r < 8; ++r)
                acc[mi][ni][r] = 0.0f;

    // Per-thread DMA geometry: thread owns 16B column-chunk ckq of rows
    // crow, crow+32, ... (X: 4 rows, W: 2 rows).
    const int crow = tid >> 3;          // 0..31
    const int ckq  = (tid & 7) << 3;    // half offset within BK

    const unsigned short* gx = xb + (size_t)(blockM + crow) * IN_F + ckq;
    const unsigned short* gw = wb + (size_t)(blockN + crow) * IN_F + ckq;
    // Low 32 bits of a generic pointer into LDS == LDS byte offset (ISA 10.2)
    const unsigned lx = (unsigned)(size_t)&lds[crow * LDSK + ckq];
    const unsigned lw = (unsigned)(size_t)&lds[XSTAGE_HALVES + crow * LDSK + ckq];

    auto copy = [&](int kt) {
        const unsigned soff = (kt & 1) ? (unsigned)STAGE_BYTES : 0u;
        const size_t   kofs = (size_t)kt * BK;   // halves
        #pragma unroll
        for (int i = 0; i < 4; ++i) {            // X tile rows: crow + 32*i
            unsigned long long ga = (unsigned long long)(gx + kofs + (size_t)i * 32 * IN_F);
            unsigned la = lx + soff + (unsigned)(i * 32 * LDSK * 2);
            asm volatile("global_load_async_to_lds_b128 %0, %1, off"
                         :: "v"(la), "v"(ga) : "memory");
        }
        #pragma unroll
        for (int i = 0; i < 2; ++i) {            // W tile rows: crow + 32*i
            unsigned long long ga = (unsigned long long)(gw + kofs + (size_t)i * 32 * IN_F);
            unsigned la = lw + soff + (unsigned)(i * 32 * LDSK * 2);
            asm volatile("global_load_async_to_lds_b128 %0, %1, off"
                         :: "v"(la), "v"(ga) : "memory");
        }
    };

    const int NKT = IN_F / BK;  // 64
    copy(0);
    #pragma unroll 1
    for (int kt = 0; kt < NKT; ++kt) {
        if (kt + 1 < NKT) {
            copy(kt + 1);
            // async loads complete in order: <=6 outstanding => stage kt landed
            asm volatile("s_wait_asynccnt 0x6" ::: "memory");
        } else {
            asm volatile("s_wait_asynccnt 0x0" ::: "memory");
        }
        __syncthreads();            // all waves' stage-kt DMA visible

        // ---- compute on stage kt (runtime offset keeps a single loop body) ----
        const unsigned so = (kt & 1) ? (unsigned)STAGE_HALVES : 0u;
        const unsigned short* Xb = lds + so;
        const unsigned short* Wb = lds + so + XSTAGE_HALVES;

        #pragma unroll
        for (int ks = 0; ks < 2; ++ks) {       // two 16x16x32 K-steps per BK=64 stage
            const int kb  = ks * 32;
            const int khA = (lane & 16) ? 8  : 0;   // A: K 0-7/16-23 vs 8-15/24-31 (ISA 7.12.2)
            const int khB = (lane & 16) ? 8  : 0;   // B: K 0-7/16-23 vs 8-15/24-31, column-major
            const int lr  = lane & 15;

            Frag a[2];
            #pragma unroll
            for (int mi = 0; mi < 2; ++mi) {
                const unsigned short* p = Xb + (waveM * 32 + mi * 16 + lr) * LDSK + kb + khA;
                a[mi].q[0] = *(const uint4*)p;          // K = kh .. kh+7
                a[mi].q[1] = *(const uint4*)(p + 16);   // K = kh+16 .. kh+23
            }
            Frag b[2];
            #pragma unroll
            for (int ni = 0; ni < 2; ++ni) {
                const unsigned short* p = Wb + (waveN * 32 + ni * 16 + lr) * LDSK + kb + khB;
                b[ni].q[0] = *(const uint4*)p;          // K = khB .. khB+7
                b[ni].q[1] = *(const uint4*)(p + 16);   // K = khB+16 .. khB+23
            }
            #pragma unroll
            for (int mi = 0; mi < 2; ++mi)
                #pragma unroll
                for (int ni = 0; ni < 2; ++ni)
                    acc[mi][ni] = __builtin_amdgcn_wmma_f32_16x16x32_bf16(
                        false, a[mi].v, false, b[ni].v,
                        (short)0, acc[mi][ni], false, false);
        }

        __syncthreads();            // reads done before stage buffer is re-filled
    }

    // C/D 16x16 f32 layout: VGPR r, lanes 0-15 -> (M=r, N=lane); lanes 16-31 -> (M=8+r, N=lane-16)
    const int mrow0 = blockM + waveM * 32 + ((lane >> 4) << 3);
    const int ncol0 = blockN + waveN * 32 + (lane & 15);
    #pragma unroll
    for (int mi = 0; mi < 2; ++mi)
        #pragma unroll
        for (int ni = 0; ni < 2; ++ni)
            #pragma unroll
            for (int r = 0; r < 8; ++r)
                out[(size_t)(mrow0 + mi * 16 + r) * OUT_F + (ncol0 + ni * 16)] = acc[mi][ni][r];
}

// ---------------------------------------------------------------------------
// Fallback: fused convert-in-kernel GEMM (used only if workspace is too small)
// ---------------------------------------------------------------------------
__global__ __launch_bounds__(256)
void lste_linear_fused(const float* __restrict__ x,
                       const signed char* __restrict__ tern,
                       const float* __restrict__ scales,
                       float* __restrict__ out) {
    __shared__ __align__(16) unsigned short Xs[BM][LDSK];
    __shared__ __align__(16) unsigned short Ws[BN][LDSK];

    const int tid   = threadIdx.x;
    const int lane  = tid & 31;
    const int wid   = tid >> 5;
    const int waveM = wid >> 1;
    const int waveN = wid & 1;
    const int blockN = blockIdx.x * BN;
    const int blockM = blockIdx.y * BM;

    v8f acc[2][2];
    #pragma unroll
    for (int mi = 0; mi < 2; ++mi)
        #pragma unroll
        for (int ni = 0; ni < 2; ++ni)
            #pragma unroll
            for (int r = 0; r < 8; ++r)
                acc[mi][ni][r] = 0.0f;

    const int NKT = IN_F / BK;
    #pragma unroll 1
    for (int kt = 0; kt < NKT; ++kt) {
        const int k0 = kt * BK;
        #pragma unroll
        for (int i = 0; i < 8; ++i) {
            int chunk = tid + i * 256;
            int row = chunk >> 4;
            int kq  = (chunk & 15) << 2;
            float4 v = *(const float4*)(x + (size_t)(blockM + row) * IN_F + k0 + kq);
            uint2 p;
            p.x = pack2bf(v.x, v.y);
            p.y = pack2bf(v.z, v.w);
            *(uint2*)&Xs[row][kq] = p;
        }
        {
            int row = tid >> 2;
            int kc  = (tid & 3) << 4;
            int4 w = *(const int4*)(tern + (size_t)(blockN + row) * IN_F + k0 + kc);
            float sc = scales[(blockN + row) * (IN_F / GS) + (k0 + kc) / GS];
            const signed char* b = (const signed char*)&w;
            unsigned q[8];
            #pragma unroll
            for (int j = 0; j < 8; ++j)
                q[j] = pack2bf((float)b[2 * j] * sc, (float)b[2 * j + 1] * sc);
            uint4* dst = (uint4*)&Ws[row][kc];
            dst[0] = make_uint4(q[0], q[1], q[2], q[3]);
            dst[1] = make_uint4(q[4], q[5], q[6], q[7]);
        }
        __syncthreads();

        #pragma unroll
        for (int ks = 0; ks < 2; ++ks) {
            const int kb  = ks * 32;
            const int khA = (lane & 16) ? 8  : 0;
            const int khB = (lane & 16) ? 8  : 0;
            const int lr  = lane & 15;

            Frag a[2];
            #pragma unroll
            for (int mi = 0; mi < 2; ++mi) {
                const unsigned short* p = &Xs[waveM * 32 + mi * 16 + lr][kb + khA];
                a[mi].q[0] = *(const uint4*)p;
                a[mi].q[1] = *(const uint4*)(p + 16);
            }
            Frag b[2];
            #pragma unroll
            for (int ni = 0; ni < 2; ++ni) {
                const unsigned short* p = &Ws[waveN * 32 + ni * 16 + lr][kb + khB];
                b[ni].q[0] = *(const uint4*)p;
                b[ni].q[1] = *(const uint4*)(p + 16);
            }
            #pragma unroll
            for (int mi = 0; mi < 2; ++mi)
                #pragma unroll
                for (int ni = 0; ni < 2; ++ni)
                    acc[mi][ni] = __builtin_amdgcn_wmma_f32_16x16x32_bf16(
                        false, a[mi].v, false, b[ni].v,
                        (short)0, acc[mi][ni], false, false);
        }
        __syncthreads();
    }

    const int mrow0 = blockM + waveM * 32 + ((lane >> 4) << 3);
    const int ncol0 = blockN + waveN * 32 + (lane & 15);
    #pragma unroll
    for (int mi = 0; mi < 2; ++mi)
        #pragma unroll
        for (int ni = 0; ni < 2; ++ni)
            #pragma unroll
            for (int r = 0; r < 8; ++r)
                out[(size_t)(mrow0 + mi * 16 + r) * OUT_F + (ncol0 + ni * 16)] = acc[mi][ni][r];
}

extern "C" void kernel_launch(void* const* d_in, const int* in_sizes, int n_in,
                              void* d_out, int out_size, void* d_ws, size_t ws_size,
                              hipStream_t stream) {
    const float*       x      = (const float*)d_in[0];
    const signed char* tern   = (const signed char*)d_in[1];   // int8 ternary
    const float*       scales = (const float*)d_in[2];
    // d_in[3] = ste_gain, d_in[4] = ste_temp : backward-only, unused in forward
    float* out = (float*)d_out;

    const size_t xb_bytes = (size_t)TOKENS * IN_F * sizeof(unsigned short);  // 64 MB
    const size_t wb_bytes = (size_t)OUT_F  * IN_F * sizeof(unsigned short);  // 32 MB

    dim3 grid(OUT_F / BN, TOKENS / BM);   // (64, 64)
    if (ws_size >= xb_bytes + wb_bytes) {
        unsigned short* xbuf = (unsigned short*)d_ws;
        unsigned short* wbuf = (unsigned short*)((char*)d_ws + xb_bytes);

        cvt_x_bf16<<<(TOKENS * IN_F) / (256 * 8), 256, 0, stream>>>(x, xbuf);
        dequant_w_bf16<<<(OUT_F * IN_F) / (256 * 16), 256, 0, stream>>>(tern, scales, wbuf);
        gemm_bf16_wmma_async<<<grid, 256, 0, stream>>>(xbuf, wbuf, out);
    } else {
        lste_linear_fused<<<grid, 256, 0, stream>>>(x, tern, scales, out);
    }
}
